// AngleHeads_28733331210488
// MI455X (gfx1250) — matrix operand order repaired and checked by reading, not compiled, not run on hardware
//
#include <hip/hip_runtime.h>
#include <math.h>

// ---------------------------------------------------------------------------
// AngleHeads for MI455X (gfx1250, wave32, WMMA).
// Expert-gather + fused per-expert MLP using v_wmma_f32_16x16x32_f16.
// ---------------------------------------------------------------------------

#define E_TYPES 20
#define NBLK    2
#define NANG    7
#define CS      384
#define CH      128
#define TILE_M  64
#define THREADS 128            // 4 waves of 32

// padded LDS strides (in halfs) to avoid bank conflicts on fragment gathers
#define AF_LD   40             // activation tile rows (32 k's used), 80 B (16B-aligned)
#define BT_LD   40             // transposed weight tile rows (32 k's used)
#define HF_LD   136            // hidden state rows (128 cols used), 272 B

typedef _Float16 half8 __attribute__((ext_vector_type(8)));
typedef _Float16 v16h  __attribute__((ext_vector_type(16)));
typedef float    v8f   __attribute__((ext_vector_type(8)));

// ---- fragment loaders (per ISA 16-bit WMMA VGPR layouts) -------------------
// A (16x32, MxK): lanes 0-15 rows M, K = {0..7, 16..23}; lanes 16-31 K = {8..15, 24..31}
__device__ __forceinline__ v16h lds_fragA(const _Float16* base, int stride, int lane) {
  const int r   = lane & 15;
  const int khi = (lane >> 4) << 3;            // 0 or 8
  const _Float16* p = base + r * stride + khi;
  half8 lo = *(const half8*)(p);               // K = khi + 0..7
  half8 hi = *(const half8*)(p + 16);          // K = 16 + khi + 0..7
  v16h f;
#pragma unroll
  for (int j = 0; j < 8; ++j) { f[j] = lo[j]; f[j + 8] = hi[j]; }
  return f;
}

// B (32x16, KxN) staged transposed in LDS as [n][k]:
// lanes 0-15 hold column n with K={0..7,16..23}; lanes 16-31 hold K={8..15,24..31}
__device__ __forceinline__ v16h lds_fragB(const _Float16* base, int stride, int lane) {
  const int n  = lane & 15;
  const int k0 = (lane >> 4) << 3;             // 0 or 8
  const _Float16* p = base + n * stride + k0;
  half8 lo = *(const half8*)(p);
  half8 hi = *(const half8*)(p + 16);
  v16h f;
#pragma unroll
  for (int j = 0; j < 8; ++j) { f[j] = lo[j]; f[j + 8] = hi[j]; }
  return f;
}

// relu * mask on 8 floats -> one 16B LDS store
__device__ __forceinline__ void cvt_store8(_Float16* dst, float4 a, float4 b, float m) {
  half8 h;
  h[0] = (_Float16)(fmaxf(a.x, 0.0f) * m);
  h[1] = (_Float16)(fmaxf(a.y, 0.0f) * m);
  h[2] = (_Float16)(fmaxf(a.z, 0.0f) * m);
  h[3] = (_Float16)(fmaxf(a.w, 0.0f) * m);
  h[4] = (_Float16)(fmaxf(b.x, 0.0f) * m);
  h[5] = (_Float16)(fmaxf(b.y, 0.0f) * m);
  h[6] = (_Float16)(fmaxf(b.z, 0.0f) * m);
  h[7] = (_Float16)(fmaxf(b.w, 0.0f) * m);
  *(half8*)dst = h;
}

// stage a 32x128 fp32 weight tile (rows k0..k0+31) transposed into Bt[n][k] f16.
// Each thread: 8 x b128 global loads, 32 x b16 LDS stores.
__device__ __forceinline__ void stage_w32(const float* __restrict__ W, int k0,
                                          _Float16* Bt_, int tid) {
#pragma unroll
  for (int seg = 0; seg < 8; ++seg) {
    const int id = (seg * THREADS + tid) * 4;  // element id in 32x128 tile
    const int kk = id >> 7;                    // row 0..31
    const int n  = id & 127;                   // col (multiple of 4)
    float4 w = *(const float4*)(W + (size_t)(k0 + kk) * CH + n);
    Bt_[(n + 0) * BT_LD + kk] = (_Float16)w.x;
    Bt_[(n + 1) * BT_LD + kk] = (_Float16)w.y;
    Bt_[(n + 2) * BT_LD + kk] = (_Float16)w.z;
    Bt_[(n + 3) * BT_LD + kk] = (_Float16)w.w;
  }
}

// ---- routing kernels -------------------------------------------------------
__global__ void route_init(int* ws) {
  if (threadIdx.x < 96) ws[threadIdx.x] = 0;   // counts[32] | fill[32] | offs[32]
}
__global__ void route_hist(const int* __restrict__ aatype, int* counts, int n) {
  int i = blockIdx.x * blockDim.x + threadIdx.x;
  if (i < n) atomicAdd(&counts[aatype[i]], 1);
}
__global__ void route_scan(int* ws) {
  if (threadIdx.x == 0) {
    int run = 0;
    for (int e = 0; e < E_TYPES; ++e) {
      ws[64 + e] = run;                        // offs
      ws[32 + e] = run;                        // fill cursor
      run += ws[e];
    }
  }
}
__global__ void route_scatter(const int* __restrict__ aatype, int* fill,
                              int* perm, int n) {
  int i = blockIdx.x * blockDim.x + threadIdx.x;
  if (i < n) {
    int p = atomicAdd(&fill[aatype[i]], 1);
    perm[p] = i;
  }
}

// ---- fused per-expert AngleHead --------------------------------------------
__global__ __launch_bounds__(THREADS)
void angle_heads_kernel(const float* __restrict__ S,
                        const float* __restrict__ Sinit,
                        const float* __restrict__ Win,  const float* __restrict__ b_in,
                        const float* __restrict__ Winit,const float* __restrict__ b_init2,
                        const float* __restrict__ Wb1,  const float* __restrict__ bb1,
                        const float* __restrict__ Wb2,  const float* __restrict__ bb2,
                        const float* __restrict__ Wout, const float* __restrict__ b_out,
                        const int* __restrict__ counts, const int* __restrict__ offs,
                        const int* __restrict__ perm,   float* __restrict__ out) {
  const int e    = blockIdx.y;
  const int tile = blockIdx.x;
  const int cnt  = counts[e];
  if (tile * TILE_M >= cnt) return;            // uniform early-exit (before any WMMA)
  const int base = offs[e];

  __shared__ int      tokIdx[TILE_M];
  __shared__ _Float16 Af[TILE_M * AF_LD];      // 5 KB   activations (relu, f16)
  __shared__ _Float16 Bt[CH * BT_LD];          // 10 KB  weight tile, transposed [n][k]
  __shared__ _Float16 Hf[TILE_M * HF_LD];      // 17 KB  relu(hidden) f16
  __shared__ float    OutS[TILE_M * 16];       // 4 KB

  const int tid  = threadIdx.x;
  const int lane = tid & 31;
  const int wv   = tid >> 5;                   // wave id 0..3
  const int arow = wv * 16;                    // this wave's 16 token rows
  const int ncol = lane & 15;
  const int mh   = ((lane >> 4) & 1) * 8;      // C/D row offset for hi half-wave

  for (int t = tid; t < TILE_M; t += THREADS) {
    int g = tile * TILE_M + t;
    tokIdx[t] = (g < cnt) ? perm[base + g] : -1;
  }
  __syncthreads();

  // hoisted, branchless activation-staging state (this thread stages one
  // half-row of 16 floats per k-step)
  const int    t_row    = tid >> 1;
  const int    c_base   = (tid & 1) * 16;
  const int    srow_raw = tokIdx[t_row];
  const float  amask    = (srow_raw >= 0) ? 1.0f : 0.0f;
  const size_t sbase    = (size_t)(srow_raw >= 0 ? srow_raw : 0) * CS + c_base;
  _Float16*    adst     = Af + t_row * AF_LD + c_base;

  v8f acc[8];
#pragma unroll
  for (int j = 0; j < 8; ++j) acc[j] = {};

  // ---- phase 1: h = relu(s) @ Win + relu(s_init) @ Winit + biases ---------
  const float* Wa = Win   + (size_t)e * CS * CH;
  const float* Wc = Winit + (size_t)e * CS * CH;
  for (int pass = 0; pass < 2; ++pass) {
    const float* act = pass ? Sinit : S;
    const float* W   = pass ? Wc : Wa;
    for (int ks = 0; ks < CS / 32; ++ks) {
      const int k0 = ks * 32;
      __syncthreads();
      { // stage activations: 4 x b128 loads, 2 x b128 LDS stores, branchless
        const float4* sp = (const float4*)(act + sbase + k0);
        float4 x0 = sp[0], x1 = sp[1], x2 = sp[2], x3 = sp[3];
        cvt_store8(adst,     x0, x1, amask);
        cvt_store8(adst + 8, x2, x3, amask);
      }
      stage_w32(W, k0, Bt, tid);
      if (ks + 1 < CS / 32)                    // prefetch next weight tile
        __builtin_prefetch(W + (size_t)(k0 + 32) * CH + tid * 32, 0, 0);
      __syncthreads();

      v16h af = lds_fragA(Af + arow * AF_LD, AF_LD, lane);
#pragma unroll
      for (int j = 0; j < 8; ++j) {
        v16h bf = lds_fragB(Bt + j * 16 * BT_LD, BT_LD, lane);
        acc[j] = __builtin_amdgcn_wmma_f32_16x16x32_f16(
            false, af, false, bf, (short)0, acc[j], false, false);
      }
    }
  }
#pragma unroll
  for (int j = 0; j < 8; ++j) {
    float bi = b_in[e * CH + j * 16 + ncol] + b_init2[e * CH + j * 16 + ncol];
#pragma unroll
    for (int r = 0; r < 8; ++r) acc[j][r] += bi;
  }
  __syncthreads();
#pragma unroll
  for (int j = 0; j < 8; ++j)
#pragma unroll
    for (int r = 0; r < 8; ++r) {
      float v = acc[j][r];
      Hf[(arow + mh + r) * HF_LD + j * 16 + ncol] = (_Float16)(v > 0.0f ? v : 0.0f);
    }
  __syncthreads();

  // ---- phase 2: residual blocks  h += relu(relu(h)@W1+b1)@W2 + b2 ---------
  for (int b = 0; b < NBLK; ++b) {
    const float* W1 = Wb1 + (size_t)(e * NBLK + b) * CH * CH;
    const float* W2 = Wb2 + (size_t)(e * NBLK + b) * CH * CH;

    v8f a_acc[8];
#pragma unroll
    for (int j = 0; j < 8; ++j) a_acc[j] = {};

    for (int ks = 0; ks < CH / 32; ++ks) {
      __syncthreads();
      stage_w32(W1, ks * 32, Bt, tid);
      __syncthreads();
      v16h af = lds_fragA(Hf + arow * HF_LD + ks * 32, HF_LD, lane);
#pragma unroll
      for (int j = 0; j < 8; ++j) {
        v16h bf = lds_fragB(Bt + j * 16 * BT_LD, BT_LD, lane);
        a_acc[j] = __builtin_amdgcn_wmma_f32_16x16x32_f16(
            false, af, false, bf, (short)0, a_acc[j], false, false);
      }
    }
    __syncthreads();                            // readers of Hf done
#pragma unroll
    for (int j = 0; j < 8; ++j) {
      float bi = bb1[(size_t)(e * NBLK + b) * CH + j * 16 + ncol];
#pragma unroll
      for (int r = 0; r < 8; ++r) {
        float v = a_acc[j][r] + bi;
        Hf[(arow + mh + r) * HF_LD + j * 16 + ncol] = (_Float16)(v > 0.0f ? v : 0.0f);
      }
    }
    __syncthreads();

    for (int ks = 0; ks < CH / 32; ++ks) {
      __syncthreads();
      stage_w32(W2, ks * 32, Bt, tid);
      __syncthreads();
      v16h af = lds_fragA(Hf + arow * HF_LD + ks * 32, HF_LD, lane);
#pragma unroll
      for (int j = 0; j < 8; ++j) {             // residual rides in C operand
        v16h bf = lds_fragB(Bt + j * 16 * BT_LD, BT_LD, lane);
        acc[j] = __builtin_amdgcn_wmma_f32_16x16x32_f16(
            false, af, false, bf, (short)0, acc[j], false, false);
      }
    }
#pragma unroll
    for (int j = 0; j < 8; ++j) {
      float bi = bb2[(size_t)(e * NBLK + b) * CH + j * 16 + ncol];
#pragma unroll
      for (int r = 0; r < 8; ++r) acc[j][r] += bi;
    }
    __syncthreads();                            // readers of Hf done
#pragma unroll
    for (int j = 0; j < 8; ++j)
#pragma unroll
      for (int r = 0; r < 8; ++r) {
        float v = acc[j][r];
        Hf[(arow + mh + r) * HF_LD + j * 16 + ncol] = (_Float16)(v > 0.0f ? v : 0.0f);
      }
    __syncthreads();
  }

  // ---- phase 3: out = relu(h) @ Wout + b_out (14 cols, padded to 16) ------
  v8f o_acc = {};
  for (int ks = 0; ks < CH / 32; ++ks) {
    __syncthreads();
    for (int i = tid; i < 16 * 32; i += THREADS) {
      int n = i >> 5, kk = i & 31;
      float v = (n < 2 * NANG)
                    ? Wout[(size_t)e * CH * (2 * NANG) + (size_t)(ks * 32 + kk) * (2 * NANG) + n]
                    : 0.0f;
      Bt[n * BT_LD + kk] = (_Float16)v;
    }
    __syncthreads();
    v16h af = lds_fragA(Hf + arow * HF_LD + ks * 32, HF_LD, lane);
    v16h bf = lds_fragB(Bt, BT_LD, lane);
    o_acc = __builtin_amdgcn_wmma_f32_16x16x32_f16(
        false, af, false, bf, (short)0, o_acc, false, false);
  }
  float bo = (ncol < 2 * NANG) ? b_out[e * (2 * NANG) + ncol] : 0.0f;
  __syncthreads();
#pragma unroll
  for (int r = 0; r < 8; ++r)
    OutS[(arow + mh + r) * 16 + ncol] = o_acc[r] + bo;
  __syncthreads();

  // ---- normalize (x,y) pairs and scatter to original token positions -----
  for (int i = tid; i < TILE_M * NANG; i += THREADS) {
    int t = i / NANG, o = i % NANG;
    int row = tokIdx[t];
    if (row >= 0) {
      float x = OutS[t * 16 + 2 * o];
      float y = OutS[t * 16 + 2 * o + 1];
      float nrm = fmaxf(sqrtf(x * x + y * y), 1e-12f);
      out[(size_t)row * (2 * NANG) + 2 * o]     = x / nrm;
      out[(size_t)row * (2 * NANG) + 2 * o + 1] = y / nrm;
    }
  }
}

// ---------------------------------------------------------------------------
extern "C" void kernel_launch(void* const* d_in, const int* in_sizes, int n_in,
                              void* d_out, int out_size, void* d_ws, size_t ws_size,
                              hipStream_t stream) {
  const float* S       = (const float*)d_in[0];
  const float* Sinit   = (const float*)d_in[1];
  const int*   aatype  = (const int*)  d_in[2];
  const float* Win     = (const float*)d_in[3];
  const float* b_in    = (const float*)d_in[4];
  const float* Winit   = (const float*)d_in[5];
  const float* b_init2 = (const float*)d_in[6];
  const float* Wb1     = (const float*)d_in[7];
  const float* bb1     = (const float*)d_in[8];
  const float* Wb2     = (const float*)d_in[9];
  const float* bb2     = (const float*)d_in[10];
  const float* Wout    = (const float*)d_in[11];
  const float* b_out   = (const float*)d_in[12];

  const int N = in_sizes[2];                    // BS * L tokens

  int* ws     = (int*)d_ws;
  int* counts = ws;                             // [32]
  int* fill   = ws + 32;                        // [32]
  int* offs   = ws + 64;                        // [32]
  int* perm   = ws + 96;                        // [N]

  route_init   <<<1, 128, 0, stream>>>(ws);
  route_hist   <<<(N + 255) / 256, 256, 0, stream>>>(aatype, counts, N);
  route_scan   <<<1, 32, 0, stream>>>(ws);
  route_scatter<<<(N + 255) / 256, 256, 0, stream>>>(aatype, fill, perm, N);

  dim3 grid((N + TILE_M - 1) / TILE_M, E_TYPES);
  angle_heads_kernel<<<grid, THREADS, 0, stream>>>(
      S, Sinit, Win, b_in, Winit, b_init2, Wb1, bb1, Wb2, bb2, Wout, b_out,
      counts, offs, perm, (float*)d_out);
}
